// DinMod_55370718380288
// MI455X (gfx1250) — matrix operand order repaired and checked by reading, not compiled, not run on hardware
//
#include <hip/hip_runtime.h>
#include <hip/hip_bf16.h>

typedef __attribute__((ext_vector_type(16))) __bf16 v16bf;
typedef __attribute__((ext_vector_type(8)))  float  v8f;

#define IN_DIM  512
#define LATENT  256
#define NPRE    204
#define NPREP   208
#define BATCH   64
#define TSTEPS  512
#define NTOK    (BATCH*TSTEPS)   // 32768
#define STATE   33
#define INTER   18
#define COMMAND 12
#define MOTOR   3

__device__ __forceinline__ float sigm(float x) { return 1.f / (1.f + __expf(-x)); }

// ---------------- prep kernels ----------------

__global__ void prep_fc1w(const float* __restrict__ w, __bf16* __restrict__ o) {
    int i = blockIdx.x * blockDim.x + threadIdx.x;
    if (i < LATENT * IN_DIM) o[i] = (__bf16)w[i];
}

// Wcat rows: [0,132) lstm_wi ; [132,150) ff1w0*mask0 ; [150,168) ff2w0*mask0 ;
//            [168,186) taw0 ; [186,204) tbw0 ; [204,208) zero pad.
// biasCat fuses the matching biases so the scan only adds recurrent terms.
__global__ void prep_wcat(const float* __restrict__ lstm_wi, const float* __restrict__ lstm_bi,
                          const float* __restrict__ ff1w0,  const float* __restrict__ ff2w0,
                          const float* __restrict__ taw0,   const float* __restrict__ tbw0,
                          const float* __restrict__ ff1b0,  const float* __restrict__ ff2b0,
                          const float* __restrict__ tab0,   const float* __restrict__ tbb0,
                          const float* __restrict__ mask0,
                          __bf16* __restrict__ Wcat, float* __restrict__ biasCat) {
    int idx = blockIdx.x * blockDim.x + threadIdx.x;
    if (idx >= NPREP * LATENT) return;
    int n = idx / LATENT, k = idx % LATENT;
    float v = 0.f, b = 0.f;
    if (n < 132)       { v = lstm_wi[n * LATENT + k];                         b = lstm_bi[n]; }
    else if (n < 150)  { int r = n - 132; v = ff1w0[r*274+k]*mask0[r*274+k];  b = ff1b0[r]; }
    else if (n < 168)  { int r = n - 150; v = ff2w0[r*274+k]*mask0[r*274+k];  b = ff2b0[r]; }
    else if (n < 186)  { int r = n - 168; v = taw0[r*274+k];                  b = tab0[r]; }
    else if (n < 204)  { int r = n - 186; v = tbw0[r*274+k];                  b = tbb0[r]; }
    Wcat[n * LATENT + k] = (__bf16)v;
    if (k == 0) biasCat[n] = b;
}

// ---------------- fused WMMA projection ----------------
// Per workgroup: rows m0..m0+15.  Stage x-panel -> LDS bf16, GEMM1 (feats, K=512)
// into LDS bf16, GEMM2 (pre = feats @ Wcat^T + bias, K=256) to global f32.
//
// Fragment layouts used here (wave32, 16-bit, contiguous k order, not the ISA table order):
//   A (16xK slice of 32): lanes 0-15 = rows, elems 0..15 -> K {0..15};
//                         lanes 16-31 shift K by +16.
//   B (Kx16 slice of 32): lanes 0-15 hold K = 0..15 (contiguous), lanes 16-31 hold K = 16..31.
//   C/D: elem e -> M = e (lanes 0-15) or e+8 (lanes 16-31); N = lane&15.
__global__ __launch_bounds__(256)
void fused_proj(const float* __restrict__ X,        // (NTOK, 512)
                const __bf16* __restrict__ Bk1,     // (256, 512) = bf16(fc1_w)
                const float* __restrict__ fc1_b,    // (256)
                const __bf16* __restrict__ Wcat,    // (208, 256)
                const float* __restrict__ biasCat,  // (208)
                float* __restrict__ pre)            // (NTOK, 208)
{
    __shared__ __bf16 aLDS[16 * IN_DIM];   // 16 KB
    __shared__ __bf16 fLDS[16 * LATENT];   //  8 KB
    const int tid   = threadIdx.x;
    const int lane  = tid & 31;
    const int wave  = tid >> 5;
    const int laneM = lane & 15;
    const int kgrpA = (lane < 16) ? 0 : 16;    // A-matrix lane->K split
    const int kgrpB = (lane < 16) ? 0 : 16;    // B-matrix lane->K split (contiguous 16)
    const int mbase = (lane < 16) ? 0 : 8;     // C/D lane->M split
    const int m0    = blockIdx.x * 16;

    for (int i = tid; i < 16 * IN_DIM; i += 256) {
        int m = i >> 9, k = i & 511;
        aLDS[i] = (__bf16)X[(size_t)(m0 + m) * IN_DIM + k];
    }
    __syncthreads();

    // GEMM1: feats(16x256) -- each wave owns two 16-col tiles
    for (int rep = 0; rep < 2; ++rep) {
        const int n0 = (wave * 2 + rep) * 16;
        v8f acc = {};
        for (int ks = 0; ks < IN_DIM / 32; ++ks) {
            const __bf16* ar = &aLDS[laneM * IN_DIM + ks * 32 + kgrpA];
            const __bf16* br = &Bk1[(size_t)(n0 + laneM) * IN_DIM + ks * 32 + kgrpB];
            v16bf a, b;
            #pragma unroll
            for (int e = 0; e < 16; ++e) a[e] = ar[e];
            #pragma unroll
            for (int e = 0; e < 16; ++e) b[e] = br[e];      // one 32B contiguous run
            acc = __builtin_amdgcn_wmma_f32_16x16x32_bf16(false, a, false, b,
                                                          (short)0, acc, false, false);
        }
        const int n = n0 + laneM;
        const float bias = fc1_b[n];
        #pragma unroll
        for (int e = 0; e < 8; ++e)
            fLDS[(mbase + e) * LATENT + n] = (__bf16)(acc[e] + bias);
    }
    __syncthreads();

    // GEMM2: pre(16x208) -- 13 tiles over 8 waves
    for (int tile = wave; tile < 13; tile += 8) {
        const int n0 = tile * 16;
        v8f acc = {};
        for (int ks = 0; ks < LATENT / 32; ++ks) {
            const __bf16* ar = &fLDS[laneM * LATENT + ks * 32 + kgrpA];
            const __bf16* br = &Wcat[(size_t)(n0 + laneM) * LATENT + ks * 32 + kgrpB];
            v16bf a, b;
            #pragma unroll
            for (int e = 0; e < 16; ++e) a[e] = ar[e];
            #pragma unroll
            for (int e = 0; e < 16; ++e) b[e] = br[e];
            acc = __builtin_amdgcn_wmma_f32_16x16x32_bf16(false, a, false, b,
                                                          (short)0, acc, false, false);
        }
        const int n = n0 + laneM;
        const float bias = biasCat[n];
        #pragma unroll
        for (int e = 0; e < 8; ++e)
            pre[(size_t)(m0 + mbase + e) * NPREP + n] = acc[e] + bias;
    }
}

// ---------------- recurrent scan: one block per batch row ----------------
__global__ __launch_bounds__(160)
void scan_kernel(const float* __restrict__ pre,
                 const float* __restrict__ lstm_wh,                         // (132,33)
                 const float* __restrict__ ff1w0, const float* __restrict__ ff2w0,
                 const float* __restrict__ taw0,  const float* __restrict__ tbw0,
                 const float* __restrict__ ff1w1, const float* __restrict__ ff2w1,
                 const float* __restrict__ taw1,  const float* __restrict__ tbw1,
                 const float* __restrict__ ff1b1, const float* __restrict__ ff2b1,
                 const float* __restrict__ tab1,  const float* __restrict__ tbb1,
                 const float* __restrict__ mask1,
                 const float* __restrict__ ff1w2, const float* __restrict__ ff2w2,
                 const float* __restrict__ taw2,  const float* __restrict__ tbw2,
                 const float* __restrict__ ff1b2, const float* __restrict__ ff2b2,
                 const float* __restrict__ tab2,  const float* __restrict__ tbb2,
                 const float* __restrict__ mask2,
                 float* __restrict__ out)
{
    __shared__ float wh[132 * 33];
    __shared__ float w1r0[18 * 18], w2r0[18 * 18], twr0[18 * 18];
    __shared__ float w1m1[12 * 30], w2m1[12 * 30], tws1[12 * 30];
    __shared__ float w1m2[3 * 15],  w2m2[3 * 15],  tws2[3 * 15];
    __shared__ float b11[12], b21[12], tb1[12];
    __shared__ float b12[3],  b22[3],  tb2[3];
    __shared__ float h[STATE], c[STATE], hl[STATE], hn[STATE];
    __shared__ float z[132];
    __shared__ float preS[NPRE];

    const int tid = threadIdx.x;
    for (int i = tid; i < 132 * 33; i += blockDim.x) wh[i] = lstm_wh[i];
    for (int i = tid; i < 18 * 18; i += blockDim.x) {
        int j = i / 18, kk = i % 18;
        w1r0[i] = ff1w0[j * 274 + 256 + kk];            // recurrent block of mask == 1
        w2r0[i] = ff2w0[j * 274 + 256 + kk];
        twr0[i] = taw0[j * 274 + 256 + kk] + tbw0[j * 274 + 256 + kk];
    }
    for (int i = tid; i < 12 * 30; i += blockDim.x) {
        float m = mask1[i];
        w1m1[i] = ff1w1[i] * m;
        w2m1[i] = ff2w1[i] * m;
        tws1[i] = taw1[i] + tbw1[i];
    }
    for (int i = tid; i < 3 * 15; i += blockDim.x) {
        float m = mask2[i];
        w1m2[i] = ff1w2[i] * m;
        w2m2[i] = ff2w2[i] * m;
        tws2[i] = taw2[i] + tbw2[i];
    }
    if (tid < 12) { b11[tid] = ff1b1[tid]; b21[tid] = ff2b1[tid]; tb1[tid] = tab1[tid] + tbb1[tid]; }
    if (tid < 3)  { b12[tid] = ff1b2[tid]; b22[tid] = ff2b2[tid]; tb2[tid] = tab2[tid] + tbb2[tid]; }
    if (tid < STATE) { h[tid] = 0.f; c[tid] = 0.f; }
    __syncthreads();

    const float* preB = pre + (size_t)blockIdx.x * TSTEPS * NPREP;
    float*       outB = out + (size_t)blockIdx.x * TSTEPS * MOTOR;

    for (int t = 0; t < TSTEPS; ++t) {
        const float* pt = preB + (size_t)t * NPREP;
        for (int i = tid; i < NPRE; i += blockDim.x) preS[i] = pt[i];
        if (t + 1 < TSTEPS) __builtin_prefetch(pt + NPREP + tid, 0, 0);  // global_prefetch
        __syncthreads();

        if (tid < 132) {                                  // z = pre + h @ wh^T
            float acc = preS[tid];
            const float* wr = &wh[tid * 33];
            #pragma unroll
            for (int i = 0; i < STATE; ++i) acc += h[i] * wr[i];
            z[tid] = acc;
        }
        __syncthreads();

        if (tid < STATE) {                                // LSTM cell
            float ia = z[tid], ig = z[33 + tid], fg = z[66 + tid], og = z[99 + tid];
            float cc = c[tid] * sigm(fg + 1.f) + tanhf(ia) * sigm(ig);
            c[tid] = cc;
            hl[tid] = tanhf(cc) * sigm(og);
        }
        __syncthreads();

        if (tid < INTER) {                                // CfC layer 0 (u-parts precomputed)
            float a1 = preS[132 + tid], a2 = preS[150 + tid];
            float at = preS[168 + tid] + preS[186 + tid];
            #pragma unroll
            for (int i = 0; i < INTER; ++i) {
                float hv = hl[i];
                a1 += hv * w1r0[tid * 18 + i];
                a2 += hv * w2r0[tid * 18 + i];
                at += hv * twr0[tid * 18 + i];
            }
            float f1 = tanhf(a1), f2 = tanhf(a2), ti = sigm(at);
            hn[tid] = f1 * (1.f - ti) + ti * f2;
        }
        __syncthreads();

        if (tid < COMMAND) {                              // CfC layer 1
            float a1 = b11[tid], a2 = b21[tid], at = tb1[tid];
            #pragma unroll
            for (int i = 0; i < 18; ++i) {
                float xv = hn[i];
                a1 += xv * w1m1[tid * 30 + i]; a2 += xv * w2m1[tid * 30 + i]; at += xv * tws1[tid * 30 + i];
            }
            #pragma unroll
            for (int i = 0; i < 12; ++i) {
                float xv = hl[18 + i];
                a1 += xv * w1m1[tid * 30 + 18 + i]; a2 += xv * w2m1[tid * 30 + 18 + i]; at += xv * tws1[tid * 30 + 18 + i];
            }
            float f1 = tanhf(a1), f2 = tanhf(a2), ti = sigm(at);
            hn[18 + tid] = f1 * (1.f - ti) + ti * f2;
        }
        __syncthreads();

        if (tid < MOTOR) {                                // CfC layer 2 -> output
            float a1 = b12[tid], a2 = b22[tid], at = tb2[tid];
            #pragma unroll
            for (int i = 0; i < 12; ++i) {
                float xv = hn[18 + i];
                a1 += xv * w1m2[tid * 15 + i]; a2 += xv * w2m2[tid * 15 + i]; at += xv * tws2[tid * 15 + i];
            }
            #pragma unroll
            for (int i = 0; i < 3; ++i) {
                float xv = hl[30 + i];
                a1 += xv * w1m2[tid * 15 + 12 + i]; a2 += xv * w2m2[tid * 15 + 12 + i]; at += xv * tws2[tid * 15 + 12 + i];
            }
            float f1 = tanhf(a1), f2 = tanhf(a2), ti = sigm(at);
            float o = f1 * (1.f - ti) + ti * f2;
            hn[30 + tid] = o;
            outB[t * MOTOR + tid] = o;
        }
        __syncthreads();

        if (tid < STATE) h[tid] = hn[tid];
        __syncthreads();
    }
}

extern "C" void kernel_launch(void* const* d_in, const int* in_sizes, int n_in,
                              void* d_out, int out_size, void* d_ws, size_t ws_size,
                              hipStream_t stream) {
    const float* x       = (const float*)d_in[0];
    const float* fc1_w   = (const float*)d_in[1];
    const float* fc1_b   = (const float*)d_in[2];
    const float* lstm_wi = (const float*)d_in[3];
    const float* lstm_bi = (const float*)d_in[4];
    const float* lstm_wh = (const float*)d_in[5];
    const float* ff1w0 = (const float*)d_in[6],  *ff2w0 = (const float*)d_in[7];
    const float* taw0  = (const float*)d_in[8],  *tbw0  = (const float*)d_in[9];
    const float* ff1b0 = (const float*)d_in[10], *ff2b0 = (const float*)d_in[11];
    const float* tab0  = (const float*)d_in[12], *tbb0  = (const float*)d_in[13];
    const float* ff1w1 = (const float*)d_in[14], *ff2w1 = (const float*)d_in[15];
    const float* taw1  = (const float*)d_in[16], *tbw1  = (const float*)d_in[17];
    const float* ff1b1 = (const float*)d_in[18], *ff2b1 = (const float*)d_in[19];
    const float* tab1  = (const float*)d_in[20], *tbb1  = (const float*)d_in[21];
    const float* ff1w2 = (const float*)d_in[22], *ff2w2 = (const float*)d_in[23];
    const float* taw2  = (const float*)d_in[24], *tbw2  = (const float*)d_in[25];
    const float* ff1b2 = (const float*)d_in[26], *ff2b2 = (const float*)d_in[27];
    const float* tab2  = (const float*)d_in[28], *tbb2  = (const float*)d_in[29];
    const float* mask0 = (const float*)d_in[30];
    const float* mask1 = (const float*)d_in[31];
    const float* mask2 = (const float*)d_in[32];

    char* ws = (char*)d_ws;
    size_t off = 0;
    float*  pre     = (float*)(ws + off); off += (size_t)NTOK * NPREP * sizeof(float);
    __bf16* Bk1     = (__bf16*)(ws + off); off += (size_t)LATENT * IN_DIM * sizeof(__bf16);
    __bf16* Wcat    = (__bf16*)(ws + off); off += (size_t)NPREP * LATENT * sizeof(__bf16);
    float*  biasCat = (float*)(ws + off); off += (size_t)NPREP * sizeof(float);
    (void)ws_size; (void)in_sizes; (void)n_in; (void)out_size;

    prep_fc1w<<<(LATENT * IN_DIM + 255) / 256, 256, 0, stream>>>(fc1_w, Bk1);
    prep_wcat<<<(NPREP * LATENT + 255) / 256, 256, 0, stream>>>(
        lstm_wi, lstm_bi, ff1w0, ff2w0, taw0, tbw0,
        ff1b0, ff2b0, tab0, tbb0, mask0, Wcat, biasCat);
    fused_proj<<<NTOK / 16, 256, 0, stream>>>(x, Bk1, fc1_b, Wcat, biasCat, pre);
    scan_kernel<<<BATCH, 160, 0, stream>>>(
        pre, lstm_wh,
        ff1w0, ff2w0, taw0, tbw0,
        ff1w1, ff2w1, taw1, tbw1, ff1b1, ff2b1, tab1, tbb1, mask1,
        ff1w2, ff2w2, taw2, tbw2, ff1b2, ff2b2, tab2, tbb2, mask2,
        (float*)d_out);
}
